// diffAttn_16243566313631
// MI455X (gfx1250) — matrix operand order repaired and checked by reading, not compiled, not run on hardware
//
#include <hip/hip_runtime.h>
#include <hip/hip_bf16.h>

typedef _Float16 half_t;
typedef __attribute__((ext_vector_type(16))) _Float16 v16h;
typedef __attribute__((ext_vector_type(8)))  _Float16 v8h;
typedef __attribute__((ext_vector_type(8)))  float    v8f;
typedef __attribute__((ext_vector_type(4)))  unsigned int u32x4;
typedef __attribute__((ext_vector_type(8)))  int      i32x8;
typedef __attribute__((ext_vector_type(4)))  int      i32x4;

#define BB 16
#define SS 1024
#define DD 512
#define HH 1024
#define EE 1024
#define CH 64
#define LAMBDA_INIT_V 0.3555090676f
#define EPS_V 1e-5f
#define SCALE_V 0.17677669529663689f   /* 1024^-0.25 */

// ---------------------------------------------------------------------------
// TDM: load a 64x32 f16 tile (row stride lda_elems) global -> LDS with
// 8-half padding after each 32-half row (matches the 40-half LDS stride).
// D# fields per CDNA5 ISA. Issued by one wave;
// completion tracked by TENSORcnt. 6-arg builtin (clang-23 toolchain).
// ---------------------------------------------------------------------------
__device__ inline void tdm_load_a_tile(const half_t* gsrc, int lda_elems,
                                       void* lds_dst, int dim0, int dim1)
{
  unsigned lds_a = (unsigned)(unsigned long long)(size_t)lds_dst;      // LDS byte offset
  unsigned long long ga = (unsigned long long)(size_t)gsrc;            // global byte addr
  u32x4 g0;
  g0[0] = 1u;                                    // count=1 (valid), user mode
  g0[1] = lds_a;                                 // lds_addr
  g0[2] = (unsigned)ga;                          // global_addr[31:0]
  g0[3] = (unsigned)((ga >> 32) & 0x01ffffffu)   // global_addr[56:32]
          | (2u << 30);                          // type=2 ("image")
  i32x8 g1;
  g1[0] = (1 << 16)    // data_size = 1 -> 2 bytes
        | (1 << 20)    // pad_enable
        | (3 << 22)    // pad_interval: 16 DWORDs (=32 halves) between pads
        | (3 << 25);   // pad_amount: 4 DWORDs (=8 halves)
  g1[1] = (int)(((unsigned)dim0 & 0xffffu) << 16);                        // tensor_dim0[15:0]
  g1[2] = (int)(((unsigned)dim0 >> 16) & 0xffffu)                         // tensor_dim0[31:16]
        | (int)(((unsigned)dim1 & 0xffffu) << 16);                        // tensor_dim1[15:0]
  g1[3] = (int)(((unsigned)dim1 >> 16) & 0xffffu) | (32 << 16);           // dim1 hi | tile_dim0=32
  g1[4] = 64;                                                             // tile_dim1=64, tile_dim2=0
  g1[5] = lda_elems;                                                      // tensor_dim0_stride[31:0]
  g1[6] = 0;
  g1[7] = 0;
  i32x4 gz4 = {0, 0, 0, 0};                      // 2-D tensor: groups 2/3 unused
  i32x8 gz8 = {0, 0, 0, 0, 0, 0, 0, 0};
  __builtin_amdgcn_tensor_load_to_lds(g0, g1, gz4, gz4, gz8, 0);
}

// ---------------------------------------------------------------------------
// Fragment helpers (CDNA5 WMMA 16x16x32 f16 layouts, CDNA5 ISA 7.12.2)
//   A tile  As[m][k]  (64 x 32, stride 40):  lane(m,lhi): k = lhi*8+{0..7},
//                                                         16+lhi*8+{0..7}
//   B tile  Bt[n][k]  (64 x 32, stride 40, TRANSPOSED at fill):
//                     lane(n,lhi): k = lhi*8+{0..7}, 16+lhi*8+{0..7}
// ---------------------------------------------------------------------------
__device__ inline v16h load_frag2(const half_t* p0, const half_t* p1) {
  v8h a = *(const v8h*)p0;
  v8h b = *(const v8h*)p1;
  v16h r;
#pragma unroll
  for (int i = 0; i < 8; ++i) { r[i] = a[i]; r[i + 8] = b[i]; }
  return r;
}

__device__ inline v16h load_afrag(const half_t* As, int m, int lhi) {
  const half_t* p = As + m * 40 + lhi * 8;
  return load_frag2(p, p + 16);
}

__device__ inline v16h load_bfrag(const half_t* Bt, int n, int lhi) {
  const half_t* p = Bt + n * 40 + lhi * 8;
  return load_frag2(p, p + 16);
}

// ---------------------------------------------------------------------------
// Generic batched GEMM: C[b] = alpha * A[b] * B[b] (+ bias[col]), f16 in,
// f32 accumulate, f32 or f16 out with arbitrary row/col strides.
// Tile 64x64x32, 128 threads = 4 waves; A tile via TDM, B tile transposed.
// ---------------------------------------------------------------------------
__global__ __launch_bounds__(128) void gemm_f16_wmma(
    const half_t* __restrict__ A, long bsA, int lda,
    const half_t* __restrict__ B, long bsB, int ldb,
    float* __restrict__ C32, half_t* __restrict__ C16,
    long bsC, long rsC, long csC,
    const float* __restrict__ bias, float alpha,
    int M, int N, int K)
{
  __shared__ __align__(16) half_t As[64 * 40];
  __shared__ __align__(16) half_t Bt[64 * 40];   // transposed: Bt[n][k]
  const int tid  = threadIdx.x;
  const int lane = tid & 31;
  const int wave = tid >> 5;
  const int lrow = lane & 15;
  const int lhi  = lane >> 4;
  const int tileN = blockIdx.x * 64;
  const int tileM = blockIdx.y * 64;
  const long bb   = blockIdx.z;
  const half_t* Ab = A + bb * bsA;
  const half_t* Bb = B + bb * bsB;

  v8f acc[4];
#pragma unroll
  for (int nt = 0; nt < 4; ++nt)
#pragma unroll
    for (int v = 0; v < 8; ++v) acc[nt][v] = 0.0f;

  const int rb0 = tid >> 3;              // 0..15  (k rows; +16 for 2nd half)
  const int cb0 = (tid & 7) * 8;         // n base

  for (int k0 = 0; k0 < K; k0 += 32) {
    // A tile 64x32 via Tensor Data Mover (wave 0 issues, TENSORcnt tracked)
    if (wave == 0) {
      tdm_load_a_tile(Ab + (long)tileM * lda + k0, lda, (void*)As, K, M);
    }
    // B tile: issue both global loads, then transpose-store to LDS
    float4 tb0 = *(const float4*)&Bb[(long)(k0 + rb0) * ldb + (tileN + cb0)];
    float4 tb1 = *(const float4*)&Bb[(long)(k0 + rb0 + 16) * ldb + (tileN + cb0)];
    const half_t* hb0 = (const half_t*)&tb0;
    const half_t* hb1 = (const half_t*)&tb1;
#pragma unroll
    for (int j = 0; j < 8; ++j) {
      Bt[(cb0 + j) * 40 + rb0]      = hb0[j];
      Bt[(cb0 + j) * 40 + rb0 + 16] = hb1[j];
    }
    if (k0 + 32 < K) {  // prefetch next B tile
      __builtin_prefetch(&Bb[(long)(k0 + 32 + (tid >> 2)) * ldb + tileN], 0, 0);
    }
    if (wave == 0) {
      __builtin_amdgcn_s_wait_tensorcnt(0);  // TDM done before barrier signal
    }
    __syncthreads();
    const v16h af = load_afrag(As, wave * 16 + lrow, lhi);
#pragma unroll
    for (int nt = 0; nt < 4; ++nt) {
      const v16h bf = load_bfrag(Bt, nt * 16 + lrow, lhi);
      acc[nt] = __builtin_amdgcn_wmma_f32_16x16x32_f16(
          false, af, false, bf, (short)0, acc[nt], false, false);
    }
    __syncthreads();
  }

  float*  Cb32 = C32 ? C32 + bb * bsC : (float*)0;
  half_t* Cb16 = C16 ? C16 + bb * bsC : (half_t*)0;
#pragma unroll
  for (int nt = 0; nt < 4; ++nt) {
    int col = tileN + nt * 16 + lrow;
    float bv = bias ? bias[col] : 0.0f;
#pragma unroll
    for (int v = 0; v < 8; ++v) {
      int row = tileM + wave * 16 + v + 8 * lhi;
      float val = acc[nt][v] * alpha + bv;
      long idx = (long)row * rsC + (long)col * csC;
      if (Cb32) Cb32[idx] = val; else Cb16[idx] = (half_t)val;
    }
  }
}

// ---------------------------------------------------------------------------
// Fused MinLSTM gate GEMM: one A (trend/seasonal) against 3 weight matrices,
// epilogue computes fp = sig(f)/(sig(f)+sig(i)), ig = sig(i)/(..)*g directly.
// ---------------------------------------------------------------------------
__global__ __launch_bounds__(128) void gate_gemm_wmma(
    const half_t* __restrict__ A, int lda,
    const half_t* __restrict__ B0, const half_t* __restrict__ B1,
    const half_t* __restrict__ B2, int ldb,
    const float* __restrict__ b0, const float* __restrict__ b1,
    const float* __restrict__ b2,
    float* __restrict__ outFp, float* __restrict__ outIg,
    int M, int N, int K)
{
  __shared__ __align__(16) half_t As[64 * 40];
  __shared__ __align__(16) half_t Bt[3][64 * 40];  // transposed: Bt[m][n][k]
  const int tid  = threadIdx.x;
  const int lane = tid & 31;
  const int wave = tid >> 5;
  const int lrow = lane & 15;
  const int lhi  = lane >> 4;
  const int tileN = blockIdx.x * 64;
  const int tileM = blockIdx.y * 64;

  v8f aF[4], aI[4], aG[4];
#pragma unroll
  for (int nt = 0; nt < 4; ++nt)
#pragma unroll
    for (int v = 0; v < 8; ++v) { aF[nt][v] = 0.f; aI[nt][v] = 0.f; aG[nt][v] = 0.f; }

  const int rb0 = tid >> 3;
  const int cb0 = (tid & 7) * 8;

  for (int k0 = 0; k0 < K; k0 += 32) {
    if (wave == 0) {
      tdm_load_a_tile(A + (long)tileM * lda + k0, lda, (void*)As, K, M);
    }
    long bo0 = (long)(k0 + rb0) * ldb + (tileN + cb0);
    long bo1 = (long)(k0 + rb0 + 16) * ldb + (tileN + cb0);
    float4 t00 = *(const float4*)&B0[bo0];
    float4 t01 = *(const float4*)&B0[bo1];
    float4 t10 = *(const float4*)&B1[bo0];
    float4 t11 = *(const float4*)&B1[bo1];
    float4 t20 = *(const float4*)&B2[bo0];
    float4 t21 = *(const float4*)&B2[bo1];
    const half_t* h00 = (const half_t*)&t00; const half_t* h01 = (const half_t*)&t01;
    const half_t* h10 = (const half_t*)&t10; const half_t* h11 = (const half_t*)&t11;
    const half_t* h20 = (const half_t*)&t20; const half_t* h21 = (const half_t*)&t21;
#pragma unroll
    for (int j = 0; j < 8; ++j) {
      int o0 = (cb0 + j) * 40 + rb0;
      Bt[0][o0]      = h00[j];
      Bt[0][o0 + 16] = h01[j];
      Bt[1][o0]      = h10[j];
      Bt[1][o0 + 16] = h11[j];
      Bt[2][o0]      = h20[j];
      Bt[2][o0 + 16] = h21[j];
    }
    if (wave == 0) {
      __builtin_amdgcn_s_wait_tensorcnt(0);
    }
    __syncthreads();
    const v16h af = load_afrag(As, wave * 16 + lrow, lhi);
#pragma unroll
    for (int nt = 0; nt < 4; ++nt) {
      const int n = nt * 16 + lrow;
      v16h bf;
      bf = load_bfrag(Bt[0], n, lhi);
      aF[nt] = __builtin_amdgcn_wmma_f32_16x16x32_f16(false, af, false, bf, (short)0, aF[nt], false, false);
      bf = load_bfrag(Bt[1], n, lhi);
      aI[nt] = __builtin_amdgcn_wmma_f32_16x16x32_f16(false, af, false, bf, (short)0, aI[nt], false, false);
      bf = load_bfrag(Bt[2], n, lhi);
      aG[nt] = __builtin_amdgcn_wmma_f32_16x16x32_f16(false, af, false, bf, (short)0, aG[nt], false, false);
    }
    __syncthreads();
  }

#pragma unroll
  for (int nt = 0; nt < 4; ++nt) {
    int col = tileN + nt * 16 + lrow;
    float bf_ = b0[col], bi_ = b1[col], bg_ = b2[col];
#pragma unroll
    for (int v = 0; v < 8; ++v) {
      int row = tileM + wave * 16 + v + 8 * lhi;
      float f  = 1.0f / (1.0f + __expf(-(aF[nt][v] + bf_)));
      float ii = 1.0f / (1.0f + __expf(-(aI[nt][v] + bi_)));
      float g  = aG[nt][v] + bg_;
      float dn = f + ii;
      long idx = (long)row * N + col;
      outFp[idx] = f / dn;
      outIg[idx] = (ii / dn) * g;
    }
  }
}

// ---------------------------------------------------------------------------
// Elementwise / scan / reduction kernels
// ---------------------------------------------------------------------------
__global__ void cvt_f32_f16(const float* __restrict__ src, half_t* __restrict__ dst, long n)
{
  long i = (long)blockIdx.x * blockDim.x + threadIdx.x;
  if (i < n) dst[i] = (half_t)src[i];
}

// dst[c*R + r] = src[r*C + c]   (src: [R,C] f32 row-major -> dst: [C,R] f16)
__global__ void transpose_cvt(const float* __restrict__ src, half_t* __restrict__ dst, int R, int C)
{
  long i = (long)blockIdx.x * blockDim.x + threadIdx.x;
  if (i >= (long)R * C) return;
  int r  = (int)(i % R);
  int cc = (int)(i / R);
  dst[i] = (half_t)src[(long)r * C + cc];
}

// xe [B,S,D] f32 -> trend_h, seas_h [B,S,D] f16; xeT_h [B,D,S] f16
__global__ void trend_seasonal(const float* __restrict__ xe,
                               half_t* __restrict__ trend_h,
                               half_t* __restrict__ seas_h,
                               half_t* __restrict__ xeT_h)
{
  long idx = (long)blockIdx.x * blockDim.x + threadIdx.x;
  int c = (int)(idx & (DD - 1));
  int s = (int)((idx >> 9) & (SS - 1));
  int b = (int)(idx >> 19);
  float x0 = xe[idx];
  float xm = (s > 0)      ? xe[idx - DD] : x0;
  float xp = (s < SS - 1) ? xe[idx + DD] : x0;
  float tr = (xm + x0 + xp) * (1.0f / 3.0f);
  trend_h[idx] = (half_t)tr;
  seas_h[idx]  = (half_t)(x0 - tr);
  xeT_h[((long)b * DD + c) * SS + s] = (half_t)x0;
}

// q scan: h_t = fp_t*h_{t-1} + ig_t, written TRANSPOSED: qT[b,h,s] (f16)
__global__ void lstm_scan_qT(const float* __restrict__ fp,
                             const float* __restrict__ ig,
                             half_t* __restrict__ qT)
{
  int t = blockIdx.x * blockDim.x + threadIdx.x;   // B*H threads
  int b = t >> 10;
  int h = t & (HH - 1);
  const float* fpb = fp + (long)b * SS * HH + h;
  const float* igb = ig + (long)b * SS * HH + h;
  half_t* dst = qT + ((long)b * HH + h) * SS;
  float hs = 0.0f;
  for (int s = 0; s < SS; s += 8) {
    __align__(16) half_t buf[8];
#pragma unroll
    for (int j = 0; j < 8; ++j) {
      long o = (long)(s + j) * HH;
      hs = fpb[o] * hs + igb[o];
      buf[j] = (half_t)hs;
    }
    *(float4*)(dst + s) = *(const float4*)buf;
  }
}

// k scan: written in natural [B,S,H] layout (f16)
__global__ void lstm_scan_k(const float* __restrict__ fp,
                            const float* __restrict__ ig,
                            half_t* __restrict__ kout)
{
  int t = blockIdx.x * blockDim.x + threadIdx.x;
  int b = t >> 10;
  int h = t & (HH - 1);
  const float* fpb = fp + (long)b * SS * HH + h;
  const float* igb = ig + (long)b * SS * HH + h;
  half_t* dst = kout + (long)b * SS * HH + h;
  float hs = 0.0f;
  for (int s = 0; s < SS; ++s) {
    long o = (long)s * HH;
    hs = fpb[o] * hs + igb[o];
    dst[o] = (half_t)hs;
  }
}

// lam = exp(sum lq1*lk1) - exp(sum lq2*lk2) + LAMBDA_INIT
__global__ void lambda_kernel(const float* __restrict__ lq1, const float* __restrict__ lk1,
                              const float* __restrict__ lq2, const float* __restrict__ lk2,
                              float* __restrict__ lam_out)
{
  __shared__ float r1[256];
  __shared__ float r2[256];
  int t = threadIdx.x;
  float s1 = 0.f, s2 = 0.f;
  for (int i = t; i < EE; i += 256) { s1 += lq1[i] * lk1[i]; s2 += lq2[i] * lk2[i]; }
  r1[t] = s1; r2[t] = s2; __syncthreads();
  for (int o = 128; o > 0; o >>= 1) {
    if (t < o) { r1[t] += r1[t + o]; r2[t] += r2[t + o]; }
    __syncthreads();
  }
  if (t == 0) *lam_out = __expf(r1[0]) - __expf(r2[0]) + LAMBDA_INIT_V;
}

// row-wise: diff = softmax(a1) - lam*softmax(a2)  (rows of 512, f16 out)
__global__ void softmax_diff(const float* __restrict__ a1,
                             const float* __restrict__ a2,
                             const float* __restrict__ lam_p,
                             half_t* __restrict__ diff)
{
  __shared__ float red[256];
  const long row = blockIdx.x;
  const int t = threadIdx.x;
  const float* r1 = a1 + row * 512;
  const float* r2 = a2 + row * 512;
  float x1a = r1[t], x1b = r1[t + 256];
  float x2a = r2[t], x2b = r2[t + 256];

  red[t] = fmaxf(x1a, x1b); __syncthreads();
  for (int o = 128; o > 0; o >>= 1) { if (t < o) red[t] = fmaxf(red[t], red[t + o]); __syncthreads(); }
  float m1 = red[0]; __syncthreads();
  float e1a = __expf(x1a - m1), e1b = __expf(x1b - m1);
  red[t] = e1a + e1b; __syncthreads();
  for (int o = 128; o > 0; o >>= 1) { if (t < o) red[t] += red[t + o]; __syncthreads(); }
  float s1 = red[0]; __syncthreads();

  red[t] = fmaxf(x2a, x2b); __syncthreads();
  for (int o = 128; o > 0; o >>= 1) { if (t < o) red[t] = fmaxf(red[t], red[t + o]); __syncthreads(); }
  float m2 = red[0]; __syncthreads();
  float e2a = __expf(x2a - m2), e2b = __expf(x2b - m2);
  red[t] = e2a + e2b; __syncthreads();
  for (int o = 128; o > 0; o >>= 1) { if (t < o) red[t] += red[t + o]; __syncthreads(); }
  float s2 = red[0];

  float lam  = *lam_p;
  float inv1 = 1.0f / s1;
  float inv2 = lam / s2;
  diff[row * 512 + t]       = (half_t)(e1a * inv1 - e2a * inv2);
  diff[row * 512 + t + 256] = (half_t)(e1b * inv1 - e2b * inv2);
}

// RMS norm over 1024 + scale by (1-lambda_init), f16 out
__global__ void rmsnorm_h(const float* __restrict__ attn, half_t* __restrict__ outh)
{
  __shared__ float red[256];
  const long row = blockIdx.x;
  const int t = threadIdx.x;
  const float4 xv = *(const float4*)(attn + row * 1024 + t * 4);
  red[t] = xv.x * xv.x + xv.y * xv.y + xv.z * xv.z + xv.w * xv.w;
  __syncthreads();
  for (int o = 128; o > 0; o >>= 1) { if (t < o) red[t] += red[t + o]; __syncthreads(); }
  float scl = rsqrtf(red[0] * (1.0f / 1024.0f) + EPS_V) * (1.0f - LAMBDA_INIT_V);
  __align__(8) half_t ob[4];
  ob[0] = (half_t)(xv.x * scl); ob[1] = (half_t)(xv.y * scl);
  ob[2] = (half_t)(xv.z * scl); ob[3] = (half_t)(xv.w * scl);
  *(float2*)(outh + row * 1024 + t * 4) = *(const float2*)ob;
}

// ---------------------------------------------------------------------------
extern "C" void kernel_launch(void* const* d_in, const int* in_sizes, int n_in,
                              void* d_out, int out_size, void* d_ws, size_t ws_size,
                              hipStream_t stream)
{
  (void)in_sizes; (void)n_in; (void)out_size; (void)ws_size;
  const float* x    = (const float*)d_in[0];
  const float* Wemb = (const float*)d_in[1];
  const float* bemb = (const float*)d_in[2];
  const float* qWf = (const float*)d_in[3];  const float* qbf = (const float*)d_in[4];
  const float* qWi = (const float*)d_in[5];  const float* qbi = (const float*)d_in[6];
  const float* qWh = (const float*)d_in[7];  const float* qbh = (const float*)d_in[8];
  const float* kWf = (const float*)d_in[9];  const float* kbf = (const float*)d_in[10];
  const float* kWi = (const float*)d_in[11]; const float* kbi = (const float*)d_in[12];
  const float* kWh = (const float*)d_in[13]; const float* kbh = (const float*)d_in[14];
  const float* Wv  = (const float*)d_in[15];
  const float* lq1 = (const float*)d_in[16]; const float* lk1 = (const float*)d_in[17];
  const float* lq2 = (const float*)d_in[18]; const float* lk2 = (const float*)d_in[19];
  const float* Wout = (const float*)d_in[20];
  float* out = (float*)d_out;

  char* wsb = (char*)d_ws;
  size_t off = 0;
  auto alloc = [&](size_t bytes) -> void* {
    void* p = wsb + off;
    off = (off + bytes + 255) & ~(size_t)255;
    return p;
  };

  const long nBSC = (long)BB * SS * CH;     // 1,048,576
  const long nBSD = (long)BB * SS * DD;     // 8,388,608
  const long nBSH = (long)BB * SS * HH;     // 16,777,216
  const long nA   = (long)BB * 512 * 512;   // 4,194,304
  const long nAttn= (long)BB * 512 * 1024;  // 8,388,608

  half_t* x_h    = (half_t*)alloc(nBSC * 2);
  half_t* Wemb_h = (half_t*)alloc((size_t)CH * DD * 2);
  half_t* qWf_h  = (half_t*)alloc((size_t)DD * HH * 2);
  half_t* qWi_h  = (half_t*)alloc((size_t)DD * HH * 2);
  half_t* qWh_h  = (half_t*)alloc((size_t)DD * HH * 2);
  half_t* kWf_h  = (half_t*)alloc((size_t)DD * HH * 2);
  half_t* kWi_h  = (half_t*)alloc((size_t)DD * HH * 2);
  half_t* kWh_h  = (half_t*)alloc((size_t)DD * HH * 2);
  half_t* WvT_h  = (half_t*)alloc((size_t)EE * EE * 2);
  half_t* WoutT_h= (half_t*)alloc((size_t)EE * EE * 2);
  float*  xe_f   = (float*)alloc(nBSD * 4);
  half_t* trend_h= (half_t*)alloc(nBSD * 2);
  half_t* seas_h = (half_t*)alloc(nBSD * 2);
  half_t* xeT_h  = (half_t*)alloc(nBSD * 2);
  float*  fp_buf = (float*)alloc(nBSH * 4);
  float*  ig_buf = (float*)alloc(nBSH * 4);
  half_t* qT_h   = (half_t*)alloc(nBSH * 2);
  half_t* k_h    = (half_t*)alloc(nBSH * 2);
  half_t* v_h    = (half_t*)alloc(nAttn * 2);
  float*  a1_f   = (float*)alloc(nA * 4);
  float*  a2_f   = (float*)alloc(nA * 4);
  half_t* diff_h = (half_t*)alloc(nA * 2);
  float*  attn_f = (float*)alloc(nAttn * 4);
  half_t* attn_h = (half_t*)alloc(nAttn * 2);
  float*  lam_p  = (float*)alloc(256);

  const dim3 b256(256);

  // 1) operand conversions
  cvt_f32_f16<<<dim3((nBSC + 255) / 256), b256, 0, stream>>>(x, x_h, nBSC);
  cvt_f32_f16<<<dim3((CH * DD + 255) / 256), b256, 0, stream>>>(Wemb, Wemb_h, (long)CH * DD);
  cvt_f32_f16<<<dim3((DD * HH + 255) / 256), b256, 0, stream>>>(qWf, qWf_h, (long)DD * HH);
  cvt_f32_f16<<<dim3((DD * HH + 255) / 256), b256, 0, stream>>>(qWi, qWi_h, (long)DD * HH);
  cvt_f32_f16<<<dim3((DD * HH + 255) / 256), b256, 0, stream>>>(qWh, qWh_h, (long)DD * HH);
  cvt_f32_f16<<<dim3((DD * HH + 255) / 256), b256, 0, stream>>>(kWf, kWf_h, (long)DD * HH);
  cvt_f32_f16<<<dim3((DD * HH + 255) / 256), b256, 0, stream>>>(kWi, kWi_h, (long)DD * HH);
  cvt_f32_f16<<<dim3((DD * HH + 255) / 256), b256, 0, stream>>>(kWh, kWh_h, (long)DD * HH);
  transpose_cvt<<<dim3((EE * EE + 255) / 256), b256, 0, stream>>>(Wv, WvT_h, EE, EE);
  transpose_cvt<<<dim3((EE * EE + 255) / 256), b256, 0, stream>>>(Wout, WoutT_h, EE, EE);
  lambda_kernel<<<dim3(1), b256, 0, stream>>>(lq1, lk1, lq2, lk2, lam_p);

  // 2) xe = x @ W_emb + b_emb   [16384 x 512], K=64
  gemm_f16_wmma<<<dim3(DD / 64, (BB * SS) / 64, 1), dim3(128), 0, stream>>>(
      x_h, 0, CH, Wemb_h, 0, DD, xe_f, (half_t*)0, 0, DD, 1,
      bemb, 1.0f, BB * SS, DD, CH);

  // 3) trend / seasonal / xe^T
  trend_seasonal<<<dim3((unsigned)(nBSD / 256)), b256, 0, stream>>>(xe_f, trend_h, seas_h, xeT_h);

  // 4) q gates (fused 3x GEMM) + scan -> qT
  gate_gemm_wmma<<<dim3(HH / 64, (BB * SS) / 64, 1), dim3(128), 0, stream>>>(
      trend_h, DD, qWf_h, qWi_h, qWh_h, HH, qbf, qbi, qbh,
      fp_buf, ig_buf, BB * SS, HH, DD);
  lstm_scan_qT<<<dim3((BB * HH) / 256), b256, 0, stream>>>(fp_buf, ig_buf, qT_h);

  // 5) k gates + scan -> k (reuse fp/ig buffers)
  gate_gemm_wmma<<<dim3(HH / 64, (BB * SS) / 64, 1), dim3(128), 0, stream>>>(
      seas_h, DD, kWf_h, kWi_h, kWh_h, HH, kbf, kbi, kbh,
      fp_buf, ig_buf, BB * SS, HH, DD);
  lstm_scan_k<<<dim3((BB * HH) / 256), b256, 0, stream>>>(fp_buf, ig_buf, k_h);

  // 6) v[b] = xe[b]^T @ Wv^T   [512 x 1024], K=1024, batched over B
  gemm_f16_wmma<<<dim3(EE / 64, 512 / 64, BB), dim3(128), 0, stream>>>(
      xeT_h, (long)DD * SS, SS, WvT_h, 0, EE, (float*)0, v_h,
      (long)512 * EE, EE, 1, (const float*)0, 1.0f, 512, EE, SS);

  // 7) a1 = q1^T @ k1 * SCALE ; a2 = q2^T @ k2 * SCALE  [512 x 512], K=1024
  gemm_f16_wmma<<<dim3(512 / 64, 512 / 64, BB), dim3(128), 0, stream>>>(
      qT_h, (long)HH * SS, SS, k_h, (long)SS * HH, HH, a1_f, (half_t*)0,
      (long)512 * 512, 512, 1, (const float*)0, SCALE_V, 512, 512, SS);
  gemm_f16_wmma<<<dim3(512 / 64, 512 / 64, BB), dim3(128), 0, stream>>>(
      qT_h + (long)512 * SS, (long)HH * SS, SS, k_h + 512, (long)SS * HH, HH,
      a2_f, (half_t*)0, (long)512 * 512, 512, 1, (const float*)0, SCALE_V, 512, 512, SS);

  // 8) diff = softmax(a1) - lam*softmax(a2)
  softmax_diff<<<dim3(BB * 512), b256, 0, stream>>>(a1_f, a2_f, lam_p, diff_h);

  // 9) attn = diff @ v   [512 x 1024], K=512
  gemm_f16_wmma<<<dim3(EE / 64, 512 / 64, BB), dim3(128), 0, stream>>>(
      diff_h, (long)512 * 512, 512, v_h, (long)512 * EE, EE, attn_f, (half_t*)0,
      (long)512 * EE, EE, 1, (const float*)0, 1.0f, 512, EE, 512);

  // 10) RMS norm * (1 - lambda_init) -> f16
  rmsnorm_h<<<dim3(BB * 512), b256, 0, stream>>>(attn_f, attn_h);

  // 11) out[b,f,c] = sum_e attn[b,c,e] * W_out[f,e]  (transposed store)
  gemm_f16_wmma<<<dim3(EE / 64, 512 / 64, BB), dim3(128), 0, stream>>>(
      attn_h, (long)512 * EE, EE, WoutT_h, 0, EE, out, (half_t*)0,
      (long)EE * 512, 1, 512, (const float*)0, 1.0f, 512, EE, SS);
}
